// TensorProductConvLayer_13657996001870
// MI455X (gfx1250) — matrix operand order repaired and checked by reading, not compiled, not run on hardware
//
#include <hip/hip_runtime.h>

#define N_NODES   10000
#define N_EDGES   160000
#define IN_DIM    56
#define EDGE_FEAT 64
#define HIDDEN    64
#define W_NUMEL   1600
#define TILE_E    64
#define NWAVE     8
#define PN  0.15811388300841897f   /* 1/sqrt(40) */
#define CG  0.5773502691896258f    /* 1/sqrt(3)  */

typedef __attribute__((ext_vector_type(16))) _Float16 v16h;
typedef __attribute__((ext_vector_type(8)))  float    v8f;

union AV { v16h v; unsigned int u[8]; uint4 q[2]; };

__device__ inline unsigned short f2h(float x) {
  union { _Float16 h; unsigned short u; } t; t.h = (_Float16)x; return t.u;
}
__device__ inline unsigned int pack2h(float a, float b) {
  union { _Float16 h[2]; unsigned int u; } t;
  t.h[0] = (_Float16)a; t.h[1] = (_Float16)b; return t.u;
}

// Async global->LDS copy of one 4B element per lane (ASYNCcnt-tracked).
__device__ inline void async_g2l_b32(void* lds_ptr, const void* gptr) {
  unsigned int lds_off = (unsigned int)(unsigned long long)lds_ptr;
  unsigned long long ga = (unsigned long long)gptr;
  asm volatile("global_load_async_to_lds_b32 %0, %1, off"
               :: "v"(lds_off), "v"(ga) : "memory");
}
__device__ inline void wait_asynccnt0() {
  asm volatile("s_wait_asynccnt 0x0" ::: "memory");
}

// A-matrix (16x32 f16) VGPR layout: lane M=l%16; VGPR j holds
// K0 = (j/4)*16 + (l/16)*8 + (j%4)*2 (+32*kstep). Source: [M][64] f16 row-major in LDS.
__device__ inline v16h load_A_lds(const unsigned short* h, int mt, int kstep, int lane) {
  const int M  = mt * 16 + (lane & 15);
  const int hb = (lane >> 4) << 3;
  AV a;
#pragma unroll
  for (int j = 0; j < 8; ++j) {
    int K0 = ((j >> 2) << 4) + hb + ((j & 3) << 1) + (kstep << 5);
    a.u[j] = *(const unsigned int*)(h + M * 64 + K0);
  }
  return a.v;
}

// B-matrix (32x16 f16) VGPR layout from pre-packed pairs wp[col][32 pairs]:
// lane N=l%16; VGPR j holds K = (j/4)*16 + (l/16)*8 + 2*(j%4) (+32*kstep).
__device__ inline v16h load_B_g(const unsigned int* wp, int c0, int kstep, int lane) {
  const int N  = lane & 15;
  const int p0 = ((lane >> 4) << 2) + (kstep << 4);
  const uint4* q = (const uint4*)(wp + (c0 + N) * 32 + p0);
  AV b; b.q[0] = q[0]; b.q[1] = q[2];
  return b.v;
}

// One 64-edge x 16-col slab of w = h @ W2 (bias added by caller per element).
__device__ inline void gemm_tile(const v16h (&A)[4][2], const unsigned int* w2p,
                                 int c0, int lane, v8f (&acc)[4]) {
#pragma unroll
  for (int mt = 0; mt < 4; ++mt) acc[mt] = (v8f){0, 0, 0, 0, 0, 0, 0, 0};
#pragma unroll
  for (int ks = 0; ks < 2; ++ks) {
    v16h b = load_B_g(w2p, c0, ks, lane);
#pragma unroll
    for (int mt = 0; mt < 4; ++mt)
      acc[mt] = __builtin_amdgcn_wmma_f32_16x16x32_f16(false, A[mt][ks], false, b,
                                                       (short)0, acc[mt], false, false);
  }
}

__global__ __launch_bounds__(256) void prep_kernel(
    const float* fc_w1, const float* fc_w2,
    unsigned int* w1p, unsigned int* w2p, float* counts, float* out)
{
  const int idx = blockIdx.x * 256 + threadIdx.x;
  if (idx < N_NODES * IN_DIM) out[idx] = 0.f;
  if (idx < N_NODES)          counts[idx] = 0.f;
  if (idx < HIDDEN * 32) {
    int n = idx >> 5, p = idx & 31;
    w1p[idx] = pack2h(fc_w1[(2 * p) * HIDDEN + n], fc_w1[(2 * p + 1) * HIDDEN + n]);
  }
  if (idx < W_NUMEL * 32) {
    int c = idx >> 5, p = idx & 31;
    w2p[idx] = pack2h(fc_w2[(2 * p) * W_NUMEL + c], fc_w2[(2 * p + 1) * W_NUMEL + c]);
  }
}

#define ESTR 72
#define EX1  32
#define ESH0 56
#define ESH1 57
#define EY1  60
#define OSTR 60

__global__ __launch_bounds__(256) void edge_kernel(
    const float* node_attr, const float* edge_attr, const float* edge_sh,
    const float* fc_b1, const float* fc_b2, const int* edge_index,
    const unsigned int* w1p, const unsigned int* w2p,
    float* counts, float* out)
{
  __shared__ unsigned short ea16[TILE_E * 64];   // edge_attr as f16
  __shared__ unsigned short h16[TILE_E * 64];    // hidden layer as f16
  __shared__ float lds_e[TILE_E * ESTR];         // per-edge x0/x1/sh0/sh1/y1
  __shared__ float lds_o[TILE_E * OSTR];         // per-edge tp accumulator (56 used)
  __shared__ int src_s[TILE_E];
  __shared__ int dst_s[TILE_E];

  const int tid  = threadIdx.x;
  const int lane = tid & 31;
  const int wv   = tid >> 5;
  const int hi   = lane >> 4;
  const int e0   = blockIdx.x * TILE_E;

  if (tid < TILE_E) {
    src_s[tid] = edge_index[e0 + tid];
    dst_s[tid] = edge_index[N_EDGES + e0 + tid];
  }
  // warm the W2 path this wave will sweep
  __builtin_prefetch(w2p + (wv << 4) * 32, 0, 3);
  __syncthreads();

  // async gather node_attr[dst] -> lds_e (per-lane addr -> per-lane LDS dest),
  // and edge_sh -> lds_e; no VGPR round-trip, overlaps with f16 conversion below.
  for (int i = tid; i < TILE_E * IN_DIM; i += 256) {     // 3584 = 14 * 256
    int e = i / IN_DIM, d = i - e * IN_DIM;
    async_g2l_b32(lds_e + e * ESTR + d, node_attr + dst_s[e] * IN_DIM + d);
  }
  for (int i = tid; i < TILE_E * 4; i += 256) {          // 256 = 1 * 256
    int e = i >> 2, q = i & 3;
    async_g2l_b32(lds_e + e * ESTR + ESH0 + q, edge_sh + (e0 + e) * 4 + q);
  }
  // edge_attr needs f32->f16 conversion: regular load/convert/store
  for (int i = tid; i < TILE_E * 64; i += 256) {
    int e = i >> 6, k = i & 63;
    ea16[i] = f2h(edge_attr[(e0 + e) * 64 + k]);
  }
  wait_asynccnt0();
  __syncthreads();

  // y1[e][u] = sum_m x1[e][u][m] * sh1[e][m]; zero tp accumulators
  for (int i = tid; i < TILE_E * 8; i += 256) {
    int e = i >> 3, u = i & 7;
    const float* E = lds_e + e * ESTR;
    lds_e[e * ESTR + EY1 + u] =
        E[EX1 + u * 3] * E[ESH1] + E[EX1 + u * 3 + 1] * E[ESH1 + 1] +
        E[EX1 + u * 3 + 2] * E[ESH1 + 2];
  }
  for (int i = tid; i < TILE_E * OSTR; i += 256) lds_o[i] = 0.f;

  // h = relu(ea @ W1 + b1): 16 [16x16] tiles over 8 waves
#pragma unroll
  for (int rep = 0; rep < 2; ++rep) {
    const int job = wv + rep * 8;
    const int mt = job >> 2, nt = job & 3;
    const int nc = nt * 16 + (lane & 15);
    const float bias = fc_b1[nc];
    v8f acc = {bias, bias, bias, bias, bias, bias, bias, bias};
#pragma unroll
    for (int ks = 0; ks < 2; ++ks) {
      v16h a = load_A_lds(ea16, mt, ks, lane);
      v16h b = load_B_g(w1p, nt * 16, ks, lane);
      acc = __builtin_amdgcn_wmma_f32_16x16x32_f16(false, a, false, b,
                                                   (short)0, acc, false, false);
    }
#pragma unroll
    for (int i = 0; i < 8; ++i) {
      int M = mt * 16 + i + (hi << 3);
      float r = acc[i] > 0.f ? acc[i] : 0.f;
      h16[M * 64 + nc] = f2h(r);
    }
  }
  __syncthreads();

  // Persistent A registers: 4 M-tiles x 2 k-steps of h
  v16h A[4][2];
#pragma unroll
  for (int mt = 0; mt < 4; ++mt)
#pragma unroll
    for (int ks = 0; ks < 2; ++ks)
      A[mt][ks] = load_A_lds(h16, mt, ks, lane);

  const int N = lane & 15;
  float s[4][8];
  v8f acc[4];

  // ---- Phase 1: w1 block (cols 0..1023, tiles 0..63) ----
  // out0[e,v] += sh0 * PN * sum_u w[u*32+v] * x0[u]; v constant per lane.
#pragma unroll
  for (int mt = 0; mt < 4; ++mt)
#pragma unroll
    for (int i = 0; i < 8; ++i) s[mt][i] = 0.f;
  for (int t = wv; t < 64; t += NWAVE) {
    const int c0 = t << 4;
    const float bias = fc_b2[c0 + N];
    gemm_tile(A, w2p, c0, lane, acc);
    const int u = c0 >> 5;
#pragma unroll
    for (int mt = 0; mt < 4; ++mt)
#pragma unroll
      for (int i = 0; i < 8; ++i) {
        const int e = mt * 16 + i + (hi << 3);
        s[mt][i] += (acc[mt][i] + bias) * lds_e[e * ESTR + u];
      }
  }
  {
    const int v = ((wv & 1) << 4) + N;
#pragma unroll
    for (int mt = 0; mt < 4; ++mt)
#pragma unroll
      for (int i = 0; i < 8; ++i) {
        const int e = mt * 16 + i + (hi << 3);
        atomicAdd(lds_o + e * OSTR + v, s[mt][i] * lds_e[e * ESTR + ESH0] * PN);
      }
  }

  // ---- Phase 2: w2 block (cols 1024..1279, tiles 64..79) ----
  // out1[e,v,m] += PN * sh1[m] * sum_u w[u*8+v] * x0[u]; v = N&7 per lane.
#pragma unroll
  for (int mt = 0; mt < 4; ++mt)
#pragma unroll
    for (int i = 0; i < 8; ++i) s[mt][i] = 0.f;
  for (int t = 64 + wv; t < 80; t += NWAVE) {
    const int c0 = t << 4;
    const float bias = fc_b2[c0 + N];
    gemm_tile(A, w2p, c0, lane, acc);
    const int u = (c0 + N - 1024) >> 3;
#pragma unroll
    for (int mt = 0; mt < 4; ++mt)
#pragma unroll
      for (int i = 0; i < 8; ++i) {
        const int e = mt * 16 + i + (hi << 3);
        s[mt][i] += (acc[mt][i] + bias) * lds_e[e * ESTR + u];
      }
  }
  {
    const int v = N & 7;
#pragma unroll
    for (int mt = 0; mt < 4; ++mt)
#pragma unroll
      for (int i = 0; i < 8; ++i) {
        const int e = mt * 16 + i + (hi << 3);
        const float* E = lds_e + e * ESTR;
        const float wq = s[mt][i] * PN;
        float* O = lds_o + e * OSTR + 32 + v * 3;
        atomicAdd(O + 0, wq * E[ESH1 + 0]);
        atomicAdd(O + 1, wq * E[ESH1 + 1]);
        atomicAdd(O + 2, wq * E[ESH1 + 2]);
      }
  }

  // ---- Phase 3: w3 block (cols 1280..1343, tiles 80..83; waves 0..3) ----
  // out1[e,v,m] += PN * sh0 * w[u*8+v] * x1[u,m]
  if (wv < 4) {
    const int c0 = (80 + wv) << 4;
    const float bias = fc_b2[c0 + N];
    gemm_tile(A, w2p, c0, lane, acc);
    const int u = (c0 + N - 1280) >> 3;
    const int v = N & 7;
#pragma unroll
    for (int mt = 0; mt < 4; ++mt)
#pragma unroll
      for (int i = 0; i < 8; ++i) {
        const int e = mt * 16 + i + (hi << 3);
        const float* E = lds_e + e * ESTR;
        const float wq = (acc[mt][i] + bias) * E[ESH0] * PN;
        float* O = lds_o + e * OSTR + 32 + v * 3;
        atomicAdd(O + 0, wq * E[EX1 + u * 3 + 0]);
        atomicAdd(O + 1, wq * E[EX1 + u * 3 + 1]);
        atomicAdd(O + 2, wq * E[EX1 + u * 3 + 2]);
      }
  }

  // ---- Phase 4: w4 block (cols 1344..1599, tiles 84..99) ----
  // out0[e,v] += CG * PN * sum_u w[u*32+v] * y1[u]; v constant per lane.
#pragma unroll
  for (int mt = 0; mt < 4; ++mt)
#pragma unroll
    for (int i = 0; i < 8; ++i) s[mt][i] = 0.f;
  for (int t = 84 + wv; t < 100; t += NWAVE) {
    const int c0 = t << 4;
    const float bias = fc_b2[c0 + N];
    gemm_tile(A, w2p, c0, lane, acc);
    const int u = (c0 - 1344) >> 5;
#pragma unroll
    for (int mt = 0; mt < 4; ++mt)
#pragma unroll
      for (int i = 0; i < 8; ++i) {
        const int e = mt * 16 + i + (hi << 3);
        s[mt][i] += (acc[mt][i] + bias) * lds_e[e * ESTR + EY1 + u];
      }
  }
  {
    const int v = ((wv & 1) << 4) + N;
#pragma unroll
    for (int mt = 0; mt < 4; ++mt)
#pragma unroll
      for (int i = 0; i < 8; ++i) {
        const int e = mt * 16 + i + (hi << 3);
        atomicAdd(lds_o + e * OSTR + v, s[mt][i] * (CG * PN));
      }
  }
  __syncthreads();

  // scatter per-edge tp into global accumulators by edge_src
  for (int i = tid; i < TILE_E * IN_DIM; i += 256) {
    int e = i / IN_DIM, d = i - e * IN_DIM;
    atomicAdd(out + src_s[e] * IN_DIM + d, lds_o[e * OSTR + d]);
  }
  if (tid < TILE_E) atomicAdd(counts + src_s[tid], 1.0f);
}

__global__ __launch_bounds__(256) void finalize_kernel(
    const float* node_attr, const float* counts, float* out)
{
  const int idx = blockIdx.x * 256 + threadIdx.x;
  if (idx >= N_NODES * IN_DIM) return;
  const int node = idx / IN_DIM;
  const float c = fmaxf(counts[node], 1.0f);
  out[idx] = out[idx] / c + node_attr[idx];
}

extern "C" void kernel_launch(void* const* d_in, const int* in_sizes, int n_in,
                              void* d_out, int out_size, void* d_ws, size_t ws_size,
                              hipStream_t stream) {
  (void)in_sizes; (void)n_in; (void)out_size; (void)ws_size;
  const float* node_attr  = (const float*)d_in[0];
  const float* edge_attr  = (const float*)d_in[1];
  const float* edge_sh    = (const float*)d_in[2];
  const float* fc_w1      = (const float*)d_in[3];
  const float* fc_b1      = (const float*)d_in[4];
  const float* fc_w2      = (const float*)d_in[5];
  const float* fc_b2      = (const float*)d_in[6];
  const int*   edge_index = (const int*)d_in[7];
  float* out = (float*)d_out;

  char* ws = (char*)d_ws;
  unsigned int* w1p = (unsigned int*)ws;                      //   8192 B
  unsigned int* w2p = (unsigned int*)(ws + 8192);             // 204800 B
  float* counts     = (float*)(ws + 8192 + 204800);           //  40000 B

  const int prep_total = N_NODES * IN_DIM;                    // covers all prep work
  prep_kernel<<<(prep_total + 255) / 256, 256, 0, stream>>>(fc_w1, fc_w2, w1p, w2p, counts, out);
  edge_kernel<<<N_EDGES / TILE_E, 256, 0, stream>>>(node_attr, edge_attr, edge_sh,
                                                    fc_b1, fc_b2, edge_index,
                                                    w1p, w2p, counts, out);
  finalize_kernel<<<(N_NODES * IN_DIM + 255) / 256, 256, 0, stream>>>(node_attr, counts, out);
}
